// PhaseActionHeads_81063212744755
// MI455X (gfx1250) — matrix operand order repaired and checked by reading, not compiled, not run on hardware
//
#include <hip/hip_runtime.h>
#include <hip/hip_bf16.h>

typedef __attribute__((ext_vector_type(16))) __bf16 v16bf;
typedef __attribute__((ext_vector_type(8)))  __bf16 v8bf;
typedef __attribute__((ext_vector_type(8)))  float  v8f;
typedef unsigned int u32x4 __attribute__((ext_vector_type(4)));
typedef unsigned int u32x8 __attribute__((ext_vector_type(8)));

// padded row length (elements) for transposed head weights: 200 bf16 = 400 B
// = 100 dwords -> LDS bank stride 36 mod 64: conflict-free across 16 n-lanes.
#define W1_LDP 200
#define LDS_B_ROWS 320   // rows of W1e^T staged in LDS by the TDM (of 384)

// ---------------------------------------------------------------- helpers

__device__ __forceinline__ float gelu_exact(float x) {
  return 0.5f * x * (1.0f + erff(x * 0.70710678118654752f));
}

__device__ __forceinline__ v8f wmma_bf16(v16bf a, v16bf b, v8f c) {
  return __builtin_amdgcn_wmma_f32_16x16x32_bf16(
      false, a, false, b, (short)0, c, false, false);
}

// A fragment (16x32 bf16, ISA 7.12.2): lane row = lane&15, half = lane>>4;
// VGPR0-3 hold K = half*8 + 0..7, VGPR4-7 hold K = 16 + half*8 + 0..7.
__device__ __forceinline__ v16bf a_frag_f32(const float* base, int ld, int k0) {
  int lane = threadIdx.x & 31;
  int row = lane & 15, lh = lane >> 4;
  const float* p0 = base + row * ld + k0 + lh * 8;
  const float* p1 = p0 + 16;
  v16bf r;
#pragma unroll
  for (int i = 0; i < 8; ++i) { r[i] = (__bf16)p0[i]; r[8 + i] = (__bf16)p1[i]; }
  return r;
}

__device__ __forceinline__ v16bf a_frag_bf(const __bf16* base, int ld, int k0) {
  int lane = threadIdx.x & 31;
  int row = lane & 15, lh = lane >> 4;
  const __bf16* p0 = base + row * ld + k0 + lh * 8;
  v8bf lo = *(const v8bf*)p0;
  v8bf hi = *(const v8bf*)(p0 + 16);
  return __builtin_shufflevector(lo, hi, 0,1,2,3,4,5,6,7,8,9,10,11,12,13,14,15);
}

// B fragment (32x16 bf16) from pre-transposed Bt[N][K]: lane col n = lane&15,
// elements 0-7 = K half*8 + 0..7, elements 8-15 = K 16 + half*8 + 0..7 (half = lane>>4).
__device__ __forceinline__ v16bf b_frag_bf(const __bf16* bt, int ld, int k0) {
  int lane = threadIdx.x & 31;
  int n = lane & 15, lh = lane >> 4;
  const __bf16* p = bt + (size_t)n * ld + k0 + lh * 8;
  v8bf lo = *(const v8bf*)p;
  v8bf hi = *(const v8bf*)(p + 16);
  return __builtin_shufflevector(lo, hi, 0,1,2,3,4,5,6,7,8,9,10,11,12,13,14,15);
}

// ---------------------------------------------------------------- prep kernels

__global__ void k_head(const float* __restrict__ ph, int* __restrict__ hi, int Bn) {
  int b = blockIdx.x * blockDim.x + threadIdx.x;
  if (b >= Bn) return;
  const float* p = ph + b * 5;
  int best = 0; float bv = p[0];
#pragma unroll
  for (int i = 1; i < 5; ++i) { if (p[i] > bv) { bv = p[i]; best = i; } }
  hi[b] = (best <= 2) ? best : 3;
}

__global__ void k_cvt_bf16(const float* __restrict__ s, __bf16* __restrict__ d, int n) {
  int i = blockIdx.x * blockDim.x + threadIdx.x;
  if (i < n) d[i] = (__bf16)s[i];
}

__global__ void k_transpose_bf16(const float* __restrict__ s, __bf16* __restrict__ d,
                                 int R, int C) {
  int i = blockIdx.x * blockDim.x + threadIdx.x;
  if (i >= R * C) return;
  int r = i / C, c = i - r * C;
  d[(size_t)c * R + r] = (__bf16)s[i];
}

// W1 [4][960][384] fp32 -> W1et [4][384][W1_LDP] bf16 (emb part, transposed, padded)
//                          W1ft [4][384][768]    bf16 (fused part, transposed)
__global__ void k_prep_w1(const float* __restrict__ W1,
                          __bf16* __restrict__ W1et, __bf16* __restrict__ W1ft) {
  int i = blockIdx.x * blockDim.x + threadIdx.x;
  const int total = 4 * 960 * 384;
  if (i >= total) return;
  int k = i / (960 * 384);
  int rem = i - k * 960 * 384;
  int io = rem / 384;
  int o = rem - io * 384;
  __bf16 v = (__bf16)W1[i];
  if (io < 192) W1et[((size_t)k * 384 + o) * W1_LDP + io] = v;
  else          W1ft[((size_t)k * 384 + o) * 768 + (io - 192)] = v;
}

// ---------------------------------------------------------------- F = fused @ W1f (all heads)
__global__ __launch_bounds__(256, 1) void k_fgemm(const __bf16* __restrict__ Abf,
                                                  const __bf16* __restrict__ Bt,
                                                  float* __restrict__ F) {
  const int m0 = blockIdx.x * 16;
  const int nh = blockIdx.y;                 // N half (0..1)
  const int w = threadIdx.x >> 5;            // 8 waves
  const int lane = threadIdx.x & 31;
  const int ln = lane & 15, lh = lane >> 4;
  const int nbase = nh * 768 + w * 96;       // 6 N-tiles per wave

  v8f acc[6];
#pragma unroll
  for (int t = 0; t < 6; ++t)
#pragma unroll
    for (int r = 0; r < 8; ++r) acc[t][r] = 0.0f;

  const __bf16* Ab = Abf + (size_t)m0 * 768;
  for (int ks = 0; ks < 24; ++ks) {          // K = 768
    v16bf a = a_frag_bf(Ab, 768, ks * 32);
#pragma unroll
    for (int t = 0; t < 6; ++t) {
      v16bf bf = b_frag_bf(Bt + (size_t)(nbase + t * 16) * 768, 768, ks * 32);
      acc[t] = wmma_bf16(a, bf, acc[t]);
    }
  }
#pragma unroll
  for (int t = 0; t < 6; ++t) {
    int n = nbase + t * 16 + ln;
#pragma unroll
    for (int r = 0; r < 8; ++r) {
      int m = m0 + r + 8 * lh;
      F[(size_t)m * 1536 + n] = acc[t][r];
    }
  }
}

// ---------------------------------------------------------------- fused mega-kernel
__global__ __launch_bounds__(256, 1) void k_main(
    const float* __restrict__ af,             // [2048][64][64]
    const unsigned char* __restrict__ mask,   // [2048][64] (bool)
    const float* __restrict__ be1,            // [192]
    const float* __restrict__ be2,            // [192]
    const float* __restrict__ b1,             // [4][384]
    const float* __restrict__ W2,             // [4][384]
    const float* __restrict__ b2,             // [4]
    const __bf16* __restrict__ We1t,          // [192][64]
    const __bf16* __restrict__ We2t,          // [192][192]
    const __bf16* __restrict__ W1et,          // [4][384][W1_LDP] padded
    const float* __restrict__ F,              // [2048][1536]
    const int* __restrict__ head_idx,         // [2048]
    float* __restrict__ out)                  // [2048][64]
{
  // TDM destination: first LDS_B_ROWS rows of the selected head's W1e^T
  __shared__ __attribute__((aligned(16))) __bf16 ldsB[LDS_B_ROWS][W1_LDP]; // 128000 B
  __shared__ __attribute__((aligned(16))) __bf16 t1[64][W1_LDP];           //  25600 B
  __shared__ float logits[64];

  const int b = blockIdx.x;
  const int tid = threadIdx.x;
  const int w = tid >> 5;           // 8 waves
  const int lane = tid & 31;
  const int mtile = w & 3;          // 4 M-tiles of 16 rows (A = 64)
  const int ngrp = w >> 2;          // N split in two halves
  const int ln = lane & 15;
  const int lh = lane >> 4;

  const int kh = head_idx[b];
  const __bf16* BtG = W1et + (size_t)kh * 384 * W1_LDP;

  if (tid < 64) logits[tid] = 0.0f;

  // ---- issue TDM: DMA LDS_B_ROWS*W1_LDP bf16 of head weights into LDS.
  // Overlaps with encoder stages below; wave 0 only (one descriptor per block).
  if (w == 0) {
    unsigned long long ga = (unsigned long long)(uintptr_t)BtG;
    unsigned int lds_addr = (unsigned int)(uintptr_t)(&ldsB[0][0]); // LDS aperture: low 32 bits
    const unsigned int tile_dw = (LDS_B_ROWS * W1_LDP * 2) / 4;     // 32000 dwords
    u32x4 g0;
    g0[0] = 1u;                                    // count=1, user mode, no gather
    g0[1] = lds_addr;                              // lds_addr
    g0[2] = (unsigned int)(ga & 0xFFFFFFFFu);      // global_addr[31:0]
    g0[3] = (unsigned int)((ga >> 32) & 0x1FFFFFFu) | (2u << 30); // addr[56:32], type=2
    u32x8 g1;
    g1[0] = (2u << 16);                            // mask=0, data_size=4B
    g1[1] = (tile_dw & 0xFFFFu) << 16;             // tensor_dim0[15:0]
    g1[2] = ((tile_dw >> 16) & 0xFFFFu) | (1u << 16); // tensor_dim0[31:16], tensor_dim1=1
    g1[3] = (tile_dw & 0xFFFFu) << 16;             // tile_dim0 (16-bit)
    g1[4] = 1u;                                    // tile_dim1=1, tile_dim2=0
    g1[5] = tile_dw;                               // tensor_dim0_stride[31:0]
    g1[6] = 0u;                                    // stride hi, dim1_stride lo
    g1[7] = 0u;
    asm volatile("tensor_load_to_lds %0, %1" :: "s"(g0), "s"(g1) : "memory");
  }

  // ---- stage 1: t1 = gelu(action_feats[b] @ We1 + be1)   [64][192], K=64
  const float* afb = af + (size_t)b * 64 * 64;
  {
    v8f acc[6];
#pragma unroll
    for (int t = 0; t < 6; ++t) {
      float bi = be1[ngrp * 96 + t * 16 + ln];
#pragma unroll
      for (int r = 0; r < 8; ++r) acc[t][r] = bi;
    }
#pragma unroll
    for (int ks = 0; ks < 2; ++ks) {
      v16bf a = a_frag_f32(afb + mtile * 16 * 64, 64, ks * 32);
#pragma unroll
      for (int t = 0; t < 6; ++t) {
        v16bf bf = b_frag_bf(We1t + (size_t)(ngrp * 96 + t * 16) * 64, 64, ks * 32);
        acc[t] = wmma_bf16(a, bf, acc[t]);
      }
    }
#pragma unroll
    for (int t = 0; t < 6; ++t) {
      int col = ngrp * 96 + t * 16 + ln;
#pragma unroll
      for (int r = 0; r < 8; ++r) {
        int row = mtile * 16 + r + 8 * lh;
        t1[row][col] = (__bf16)gelu_exact(acc[t][r]);
      }
    }
  }
  __syncthreads();

  // ---- stage 2: emb = gelu(t1 @ We2 + be2), written back in place into t1
  {
    v8f acc[6];
#pragma unroll
    for (int t = 0; t < 6; ++t) {
      float bi = be2[ngrp * 96 + t * 16 + ln];
#pragma unroll
      for (int r = 0; r < 8; ++r) acc[t][r] = bi;
    }
#pragma unroll
    for (int ks = 0; ks < 6; ++ks) {
      v16bf a = a_frag_bf(&t1[mtile * 16][0], W1_LDP, ks * 32);
#pragma unroll
      for (int t = 0; t < 6; ++t) {
        v16bf bf = b_frag_bf(We2t + (size_t)(ngrp * 96 + t * 16) * 192, 192, ks * 32);
        acc[t] = wmma_bf16(a, bf, acc[t]);
      }
    }
    __syncthreads();   // all stage-2 reads of t1 complete before overwrite
#pragma unroll
    for (int t = 0; t < 6; ++t) {
      int col = ngrp * 96 + t * 16 + ln;
#pragma unroll
      for (int r = 0; r < 8; ++r) {
        int row = mtile * 16 + r + 8 * lh;
        t1[row][col] = (__bf16)gelu_exact(acc[t][r]);
      }
    }
  }
  // head weights must have landed in LDS before stage 3 reads them
  if (w == 0) __builtin_amdgcn_s_wait_tensorcnt(0);
  __syncthreads();

  // ---- stage 3: selected head. h = gelu(emb @ W1e[k] + (F[b,k] + b1[k]))
  const float* Fb  = F  + (size_t)b * 1536 + (size_t)kh * 384;
  const float* b1k = b1 + kh * 384;
  const float* W2k = W2 + kh * 384;

  v8f acc[12];                                // wave covers 192 of N=384
#pragma unroll
  for (int t = 0; t < 12; ++t) {
    int col = ngrp * 192 + t * 16 + ln;
    float ci = Fb[col] + b1k[col];
#pragma unroll
    for (int r = 0; r < 8; ++r) acc[t][r] = ci;
  }
#pragma unroll
  for (int ks = 0; ks < 6; ++ks) {            // K = 192
    v16bf a = a_frag_bf(&t1[mtile * 16][0], W1_LDP, ks * 32);
#pragma unroll
    for (int t = 0; t < 12; ++t) {
      int nrow = ngrp * 192 + t * 16;
      v16bf bf;
      if (nrow + 16 <= LDS_B_ROWS)
        bf = b_frag_bf(&ldsB[nrow][0], W1_LDP, ks * 32);        // ds_load path
      else
        bf = b_frag_bf(BtG + (size_t)nrow * W1_LDP, W1_LDP, ks * 32); // tail from L2
      acc[t] = wmma_bf16(a, bf, acc[t]);
    }
  }

  // gelu + dot with W2[k]; reduce n in-lane, then across 16-lane groups
  float p[8];
#pragma unroll
  for (int r = 0; r < 8; ++r) p[r] = 0.0f;
#pragma unroll
  for (int t = 0; t < 12; ++t) {
    float w2v = W2k[ngrp * 192 + t * 16 + ln];
#pragma unroll
    for (int r = 0; r < 8; ++r) p[r] += gelu_exact(acc[t][r]) * w2v;
  }
#pragma unroll
  for (int r = 0; r < 8; ++r) {
    float v = p[r];
    v += __shfl_xor(v, 1, 32);
    v += __shfl_xor(v, 2, 32);
    v += __shfl_xor(v, 4, 32);
    v += __shfl_xor(v, 8, 32);
    if (ln == 0) atomicAdd(&logits[mtile * 16 + r + 8 * lh], v);
  }
  __syncthreads();

  if (tid < 64) {
    float v = logits[tid] + b2[kh];
    out[(size_t)b * 64 + tid] = mask[(size_t)b * 64 + tid] ? -10000.0f : v;
  }
}

// ---------------------------------------------------------------- launch

extern "C" void kernel_launch(void* const* d_in, const int* in_sizes, int n_in,
                              void* d_out, int out_size, void* d_ws, size_t ws_size,
                              hipStream_t stream) {
  (void)in_sizes; (void)n_in; (void)out_size; (void)ws_size;
  const float*         af    = (const float*)d_in[0];
  const unsigned char* mask  = (const unsigned char*)d_in[1];
  const float*         fused = (const float*)d_in[2];
  const float*         phase = (const float*)d_in[3];
  const float*         We1   = (const float*)d_in[4];
  const float*         be1   = (const float*)d_in[5];
  const float*         We2   = (const float*)d_in[6];
  const float*         be2   = (const float*)d_in[7];
  const float*         W1    = (const float*)d_in[8];
  const float*         b1    = (const float*)d_in[9];
  const float*         W2    = (const float*)d_in[10];
  const float*         b2    = (const float*)d_in[11];
  float* out = (float*)d_out;

  // workspace layout (256B-aligned)
  char* ws = (char*)d_ws;
  size_t off = 0;
  auto take = [&](size_t bytes) -> char* {
    char* p = ws + off;
    off = (off + bytes + 255) & ~(size_t)255;
    return p;
  };
  int*    head_idx = (int*)   take((size_t)2048 * 4);
  __bf16* We1t     = (__bf16*)take((size_t)192 * 64 * 2);
  __bf16* We2t     = (__bf16*)take((size_t)192 * 192 * 2);
  __bf16* W1et     = (__bf16*)take((size_t)4 * 384 * W1_LDP * 2);
  __bf16* W1ft     = (__bf16*)take((size_t)4 * 384 * 768 * 2);
  __bf16* fusedbf  = (__bf16*)take((size_t)2048 * 768 * 2);
  float*  F        = (float*) take((size_t)2048 * 1536 * 4);

  // prep
  k_head<<<(2048 + 255) / 256, 256, 0, stream>>>(phase, head_idx, 2048);
  k_transpose_bf16<<<(64 * 192 + 255) / 256, 256, 0, stream>>>(We1, We1t, 64, 192);
  k_transpose_bf16<<<(192 * 192 + 255) / 256, 256, 0, stream>>>(We2, We2t, 192, 192);
  k_prep_w1<<<(4 * 960 * 384 + 255) / 256, 256, 0, stream>>>(W1, W1et, W1ft);
  k_cvt_bf16<<<(2048 * 768 + 255) / 256, 256, 0, stream>>>(fused, fusedbf, 2048 * 768);

  // F = fused @ W1_fused for all 4 heads  (M=2048, N=1536, K=768)
  k_fgemm<<<dim3(2048 / 16, 2), 256, 0, stream>>>(fusedbf, W1ft, F);

  // fused encoder + selected-head MLP + routing + mask (TDM-staged weights)
  k_main<<<2048, 256, 0, stream>>>(af, mask, be1, be2, b1, W2, b2,
                                   We1t, We2t, W1et, F, head_idx, out);
}
